// Retention_87170656240221
// MI455X (gfx1250) — matrix operand order repaired and checked by reading, not compiled, not run on hardware
//
#include <hip/hip_runtime.h>
#include <hip/hip_bf16.h>

// ---------------------------------------------------------------------------
// Retention forward for MI455X (gfx1250, wave32, WMMA).
//   K0: one-shot f32 -> bf16 downconvert of x and W (off the hot path)
//   K1: qkv = x @ W^T + b        -> q,k (bf16 row-major), v (bf16 TRANSPOSED)
//       pure bf16 b128 loads + v_wmma, 2 n-tiles per wave (A-frag reuse)
//   K2: S   = mask(q @ k^T)/sqrt(d) * alpha^(i-j)   (causal tiles only, bf16)
//   K3: out = S @ v              (f32 output)
// All GEMMs: v_wmma_f32_16x16x32_bf16, f32 accumulate.
// ---------------------------------------------------------------------------

#define EMBED_DIM 768
#define SEQ_LEN   1024
#define BATCH     16
#define QKV_DIM   2304

#define LN_ALPHA   (-0.01005033585350145f)   // ln(0.99)
#define INV_SQRT_D (0.03608439182435161f)    // 1/sqrt(768)

typedef __attribute__((ext_vector_type(16))) __bf16    v16bf;
typedef __attribute__((ext_vector_type(8)))  float     v8f;
typedef __attribute__((ext_vector_type(4)))  unsigned  u32x4;
typedef __attribute__((ext_vector_type(4)))  float     f32x4;

union FragAB { v16bf v; u32x4 q[2]; unsigned u[8]; };
union FragC  { v8f   v; float  f[8]; };

__device__ __forceinline__ unsigned pk_bf16(float a, float b) {
    unsigned ua = __builtin_bit_cast(unsigned, a);
    unsigned ub = __builtin_bit_cast(unsigned, b);
    ua += 0x7FFFu + ((ua >> 16) & 1u);      // round-to-nearest-even
    ub += 0x7FFFu + ((ub >> 16) & 1u);
    return (ua >> 16) | (ub & 0xFFFF0000u);
}

__device__ __forceinline__ v8f wmma_bf16(const FragAB& a, const FragAB& b, v8f c) {
    return __builtin_amdgcn_wmma_f32_16x16x32_bf16(false, a.v, false, b.v,
                                                   (short)0, c, false, false);
}

// A fragment (16x32 bf16, MxK): lane = {row = lane&15, half = lane>>4}
// VGPR0..3 hold K = half*8 + 0..7, VGPR4..7 hold K = 16 + half*8 + 0..7.
__device__ __forceinline__ void load_a_bf16(FragAB& fr, const __hip_bfloat16* row_base,
                                            int k, int half) {
    const __hip_bfloat16* p = row_base + k + half * 8;
    fr.q[0] = *(const u32x4*)(p);
    fr.q[1] = *(const u32x4*)(p + 16);
}

// B fragment (32x16 bf16, KxN): lane = {col = lane&15, half = lane>>4}
// VGPR0..3 hold K = half*8 + 0..7, VGPR4..7 hold K = 16 + half*8 + 0..7.
__device__ __forceinline__ void load_b_bf16(FragAB& fr, const __hip_bfloat16* col_base,
                                            int k, int half) {
    const __hip_bfloat16* p = col_base + k + half * 8;
    fr.q[0] = *(const u32x4*)(p);
    fr.q[1] = *(const u32x4*)(p + 16);
}

// ---------------------------------------------------------------------------
// K0: elementwise f32 -> bf16, 8 elements per thread, b128 in / b128 out.
// ---------------------------------------------------------------------------
__global__ __launch_bounds__(256) void cvt_bf16_kernel(
    const float* __restrict__ src, __hip_bfloat16* __restrict__ dst, int n8)
{
    const int i = blockIdx.x * 256 + threadIdx.x;
    if (i >= n8) return;
    const f32x4* p = (const f32x4*)src + 2 * (size_t)i;
    f32x4 a = p[0];
    f32x4 b = p[1];
    unsigned o[4] = { pk_bf16(a.x, a.y), pk_bf16(a.z, a.w),
                      pk_bf16(b.x, b.y), pk_bf16(b.z, b.w) };
    *((u32x4*)dst + (size_t)i) = *(u32x4*)o;
}

// ---------------------------------------------------------------------------
// K1: QKV projection. C[16384,2304] = x[16384,768] @ W^T + b.
// grid = (1024 m-tiles, 9 n-groups), block = 256 (8 waves, 2 tiles per wave).
// ---------------------------------------------------------------------------
__device__ __forceinline__ void store_qkv_tile(
    float* L, FragC& acc, int lane, int m0, int n0,
    const float* __restrict__ bias,
    __hip_bfloat16* __restrict__ qw, __hip_bfloat16* __restrict__ kw,
    __hip_bfloat16* __restrict__ vT)
{
    const int half = lane >> 4;
    const int rc   = lane & 15;

    const float bv = bias[n0 + rc];   // C layout: col = lane&15 for all 8 rows
    #pragma unroll
    for (int r = 0; r < 8; ++r) acc.f[r] += bv;

    #pragma unroll
    for (int r = 0; r < 8; ++r) L[(r + half * 8) * 16 + rc] = acc.f[r];
    __builtin_amdgcn_wave_barrier();

    const int row = lane >> 1;
    const int c0  = (lane & 1) * 8;

    if (n0 < 2 * EMBED_DIM) {
        // q / k: bf16 row-major [BATCH*SEQ, EMBED]
        __hip_bfloat16* dst = (n0 < EMBED_DIM) ? qw : kw;
        const int nn = (n0 < EMBED_DIM) ? n0 : n0 - EMBED_DIM;
        unsigned o[4];
        #pragma unroll
        for (int t = 0; t < 4; ++t)
            o[t] = pk_bf16(L[row * 16 + c0 + 2 * t], L[row * 16 + c0 + 2 * t + 1]);
        *(u32x4*)(dst + (size_t)(m0 + row) * EMBED_DIM + nn + c0) = *(u32x4*)o;
    } else {
        // v: bf16 TRANSPOSED per batch: vT[b][d][s]. Lane covers one d-row,
        // 8 consecutive s -> contiguous 16B store.
        const int nn   = n0 - 2 * EMBED_DIM;
        const int bidx = m0 >> 10;            // m0 / SEQ_LEN
        const int s0   = m0 & (SEQ_LEN - 1);
        unsigned o[4];
        #pragma unroll
        for (int t = 0; t < 4; ++t)
            o[t] = pk_bf16(L[(c0 + 2 * t) * 16 + row], L[(c0 + 2 * t + 1) * 16 + row]);
        *(u32x4*)(vT + ((size_t)bidx * EMBED_DIM + nn + row) * SEQ_LEN + s0 + c0) =
            *(u32x4*)o;
    }
    __builtin_amdgcn_wave_barrier();   // LDS region reused by next tile
}

__global__ __launch_bounds__(256) void qkv_kernel(
    const __hip_bfloat16* __restrict__ xb, const __hip_bfloat16* __restrict__ Wb,
    const float* __restrict__ bias,
    __hip_bfloat16* __restrict__ qw, __hip_bfloat16* __restrict__ kw,
    __hip_bfloat16* __restrict__ vT)
{
    __shared__ float lds[8][256];
    const int wave = threadIdx.x >> 5;
    const int lane = threadIdx.x & 31;
    const int half = lane >> 4;
    const int rc   = lane & 15;

    const int m0 = blockIdx.x * 16;
    const int n0 = (blockIdx.y * 16 + wave * 2) * 16;   // two adjacent n-tiles
    const int n1 = n0 + 16;

    const __hip_bfloat16* arow = xb + (size_t)(m0 + rc) * EMBED_DIM;
    const __hip_bfloat16* b0   = Wb + (size_t)(n0 + rc) * EMBED_DIM;
    const __hip_bfloat16* b1   = Wb + (size_t)(n1 + rc) * EMBED_DIM;

    FragC acc0, acc1;
    #pragma unroll
    for (int r = 0; r < 8; ++r) { acc0.f[r] = 0.f; acc1.f[r] = 0.f; }

    #pragma unroll 4
    for (int k = 0; k < EMBED_DIM; k += 32) {
        FragAB a, bb0, bb1;
        load_a_bf16(a, arow, k, half);       // shared A-frag for both tiles
        load_b_bf16(bb0, b0, k, half);
        load_b_bf16(bb1, b1, k, half);
        acc0.v = wmma_bf16(a, bb0, acc0.v);
        acc1.v = wmma_bf16(a, bb1, acc1.v);
    }

    float* L = lds[wave];
    store_qkv_tile(L, acc0, lane, m0, n0, bias, qw, kw, vT);
    store_qkv_tile(L, acc1, lane, m0, n1, bias, qw, kw, vT);
}

// ---------------------------------------------------------------------------
// K2: causal decayed scores. S[b][i][j] = (q_i . k_j)/sqrt(d) * alpha^(i-j).
// grid = (16 batches, 64 q-tiles); 8 waves stride the causal k-tiles.
// Even-diagonal partner tile is zero-filled so K3 can read full K=32 chunks.
// ---------------------------------------------------------------------------
__global__ __launch_bounds__(256) void score_kernel(
    const __hip_bfloat16* __restrict__ qw, const __hip_bfloat16* __restrict__ kw,
    __hip_bfloat16* __restrict__ S)
{
    __shared__ float lds[8][256];
    const int wave = threadIdx.x >> 5;
    const int lane = threadIdx.x & 31;
    const int half = lane >> 4;
    const int rc   = lane & 15;
    const int b    = blockIdx.x;
    const int qt   = blockIdx.y;
    const int i0   = qt * 16;

    const __hip_bfloat16* qrow = qw + ((size_t)b * SEQ_LEN + i0 + rc) * EMBED_DIM;
    __hip_bfloat16* Sb = S + (size_t)b * SEQ_LEN * SEQ_LEN;

    const int row = lane >> 1;
    const int c0  = (lane & 1) * 8;
    float* L = lds[wave];

    for (int kt = wave; kt <= qt; kt += 8) {
        const int j0 = kt * 16;
        // B operand is k^T: lane's column = key row j0+rc, contiguous along d.
        const __hip_bfloat16* kcol =
            kw + ((size_t)b * SEQ_LEN + j0 + rc) * EMBED_DIM;

        FragC acc;
        #pragma unroll
        for (int r = 0; r < 8; ++r) acc.f[r] = 0.f;

        #pragma unroll 4
        for (int k = 0; k < EMBED_DIM; k += 32) {
            FragAB a, bb;
            load_a_bf16(a, qrow, k, half);
            load_b_bf16(bb, kcol, k, half);
            acc.v = wmma_bf16(a, bb, acc.v);
        }

        // decay mask * 1/sqrt(d), elementwise on the C layout
        const int j = j0 + rc;
        #pragma unroll
        for (int r = 0; r < 8; ++r) {
            const int i    = i0 + r + half * 8;
            const int diff = i - j;
            const float sc =
                (diff >= 0) ? __expf(LN_ALPHA * (float)diff) * INV_SQRT_D : 0.f;
            acc.f[r] *= sc;
        }

        #pragma unroll
        for (int r = 0; r < 8; ++r) L[(r + half * 8) * 16 + rc] = acc.f[r];
        __builtin_amdgcn_wave_barrier();

        unsigned o[4];
        #pragma unroll
        for (int t = 0; t < 4; ++t)
            o[t] = pk_bf16(L[row * 16 + c0 + 2 * t], L[row * 16 + c0 + 2 * t + 1]);
        *(u32x4*)(Sb + (size_t)(i0 + row) * SEQ_LEN + j0 + c0) = *(u32x4*)o;
        __builtin_amdgcn_wave_barrier();

        // zero the just-above-diagonal tile when qt is even (K=32 chunk pad)
        if (kt == qt && !(qt & 1) && (qt + 1) < (SEQ_LEN / 16)) {
            u32x4 z = {0u, 0u, 0u, 0u};
            *(u32x4*)(Sb + (size_t)(i0 + row) * SEQ_LEN + (j0 + 16) + c0) = z;
        }
    }
}

// ---------------------------------------------------------------------------
// K3: out = S @ V using vT.  grid = (16 batches, 64 q-tiles, 6 n-groups),
// wave owns one 16x16 out tile, loops K=32 chunks of causal keys.
// ---------------------------------------------------------------------------
__global__ __launch_bounds__(256) void out_kernel(
    const __hip_bfloat16* __restrict__ S, const __hip_bfloat16* __restrict__ vT,
    float* __restrict__ out)
{
    __shared__ float lds[8][256];
    const int wave = threadIdx.x >> 5;
    const int lane = threadIdx.x & 31;
    const int half = lane >> 4;
    const int rc   = lane & 15;
    const int b    = blockIdx.x;
    const int qt   = blockIdx.y;
    const int n0   = (blockIdx.z * 8 + wave) * 16;
    const int i0   = qt * 16;

    const __hip_bfloat16* Srow = S  + ((size_t)b * SEQ_LEN + i0 + rc) * SEQ_LEN;
    const __hip_bfloat16* Vcol = vT + ((size_t)b * EMBED_DIM + n0 + rc) * SEQ_LEN;

    FragC acc;
    #pragma unroll
    for (int r = 0; r < 8; ++r) acc.f[r] = 0.f;

    const int nchunks = qt / 2 + 1;   // ceil((qt+1)*16 / 32), pad tile zeroed in K2
    for (int t = 0; t < nchunks; ++t) {
        const int j = t * 32;
        FragAB a, bb;
        load_a_bf16(a, Srow, j, half);   // S row-major: perfect A-fragment
        load_b_bf16(bb, Vcol, j, half);  // vT row-major: perfect B-fragment
        acc.v = wmma_bf16(a, bb, acc.v);
    }

    float* L = lds[wave];
    #pragma unroll
    for (int r = 0; r < 8; ++r) L[(r + half * 8) * 16 + rc] = acc.f[r];
    __builtin_amdgcn_wave_barrier();

    const int row = lane >> 1;
    const int c0  = (lane & 1) * 8;
    float* op = out + ((size_t)b * SEQ_LEN + i0 + row) * EMBED_DIM + n0 + c0;
    f32x4 o0 = *(f32x4*)&L[row * 16 + c0];
    f32x4 o1 = *(f32x4*)&L[row * 16 + c0 + 4];
    *(f32x4*)(op)     = o0;
    *(f32x4*)(op + 4) = o1;
}

// ---------------------------------------------------------------------------
extern "C" void kernel_launch(void* const* d_in, const int* in_sizes, int n_in,
                              void* d_out, int out_size, void* d_ws, size_t ws_size,
                              hipStream_t stream) {
    const float* x    = (const float*)d_in[0];   // [16,1024,768]
    const float* W    = (const float*)d_in[1];   // [2304,768]
    const float* bias = (const float*)d_in[2];   // [2304]
    float* out        = (float*)d_out;           // [16,1024,768]

    char* ws = (char*)d_ws;
    const size_t XN  = (size_t)BATCH * SEQ_LEN * EMBED_DIM;      // 12,582,912
    const size_t WN  = (size_t)QKV_DIM * EMBED_DIM;              //  1,769,472
    const size_t QK  = XN * sizeof(__hip_bfloat16);

    __hip_bfloat16* xb = (__hip_bfloat16*)(ws);
    __hip_bfloat16* Wb = (__hip_bfloat16*)(ws + QK);
    __hip_bfloat16* qw = (__hip_bfloat16*)(ws + QK + WN * 2);
    __hip_bfloat16* kw = (__hip_bfloat16*)(ws + 2 * QK + WN * 2);
    __hip_bfloat16* vT = (__hip_bfloat16*)(ws + 3 * QK + WN * 2);
    __hip_bfloat16* S  = (__hip_bfloat16*)(ws + 4 * QK + WN * 2); // [16,1024,1024]

    const int xn8 = (int)(XN / 8);
    const int wn8 = (int)(WN / 8);
    cvt_bf16_kernel<<<(xn8 + 255) / 256, 256, 0, stream>>>(x, xb, xn8);
    cvt_bf16_kernel<<<(wn8 + 255) / 256, 256, 0, stream>>>(W, Wb, wn8);

    qkv_kernel  <<<dim3(1024, 9),    256, 0, stream>>>(xb, Wb, bias, qw, kw, vT);
    score_kernel<<<dim3(16, 64),     256, 0, stream>>>(qw, kw, S);
    out_kernel  <<<dim3(16, 64, 6),  256, 0, stream>>>(S, vT, out);
}
